// SelectiveScan_41936060678335
// MI455X (gfx1250) — matrix operand order repaired and checked by reading, not compiled, not run on hardware
//
#include <hip/hip_runtime.h>

// ---------------------------------------------------------------------------
// Selective scan (Mamba) for MI455X / gfx1250, wave32, WMMA bf16.
//   1) cvt: f32 -> bf16 for x, Wx, Wdt, WB, WC, Wout
//   2) gemm_bf16_wmma<EPI>: block tile 64M x 256N (8 waves, wave = 32M x 64N),
//      A tile staged to LDS via global_load_async_to_lds_b128 (double buffered,
//      padded stride), B streamed from global with grouped b128 loads.
//   3) scan: wave = 4 channels; lane = (channel quarter, 4 states).
//   4) out-proj GEMM (f32 to d_out).
// ---------------------------------------------------------------------------

typedef __attribute__((ext_vector_type(16))) __bf16 v16bf;
typedef __attribute__((ext_vector_type(8)))  __bf16 v8bf;
typedef __attribute__((ext_vector_type(4)))  __bf16 v4bf;
typedef __attribute__((ext_vector_type(8)))  float  v8f;
typedef __attribute__((ext_vector_type(4)))  float  v4f;
typedef int v4i_vs __attribute__((vector_size(16)));   // matches builtin param type

#define D_MODEL 512
#define D_STATE 32
#define D_INNER 1024
#define BATCH   2
#define SEQ_LEN 1024
#define MTOT    (BATCH * SEQ_LEN)          // 2048
#define NBC     (D_INNER * D_STATE)        // 32768

#define LDS_STRIDE 40   // elements (80B): 16 parallel rows hit distinct LDS banks

#if __has_builtin(__builtin_amdgcn_global_load_async_to_lds_b128) && \
    __has_builtin(__builtin_amdgcn_s_wait_asynccnt)
#define USE_ASYNC_LDS 1
#else
#define USE_ASYNC_LDS 0
#endif

// 16-byte global -> LDS copy (async path if the toolchain exposes it)
__device__ __forceinline__ void g2l_16B(__bf16* lds_dst, const __bf16* gsrc)
{
#if USE_ASYNC_LDS
    __builtin_amdgcn_global_load_async_to_lds_b128(
        (__attribute__((address_space(1))) v4i_vs*)gsrc,
        (__attribute__((address_space(3))) v4i_vs*)lds_dst,
        /*offset=*/0, /*cpol=*/0);
#else
    *reinterpret_cast<v8bf*>(lds_dst) = *reinterpret_cast<const v8bf*>(gsrc);
#endif
}

__device__ __forceinline__ void stage_wait_and_barrier()
{
#if USE_ASYNC_LDS
    __builtin_amdgcn_s_wait_asynccnt(0);
#endif
    __syncthreads();
}

// ---------------------------------------------------------------- convert ---
__global__ __launch_bounds__(256) void ss_cvt_f32_to_bf16(
    const float* __restrict__ in, __bf16* __restrict__ out, int n)
{
    int i = (blockIdx.x * 256 + threadIdx.x) * 4;
    if (i >= n) return;
    v4f v = *reinterpret_cast<const v4f*>(in + i);
    v4bf o;
    o.x = (__bf16)v.x; o.y = (__bf16)v.y; o.z = (__bf16)v.z; o.w = (__bf16)v.w;
    *reinterpret_cast<v4bf*>(out + i) = o;
}

// ------------------------------------------------------------------- GEMM ---
// Block = 256 threads = 8 waves as 2(M) x 4(N). Block tile 64M x 256N.
// Wave tile 32M x 64N = 8 WMMA accumulators. K in steps of 32.
template <int EPI>
__global__ __launch_bounds__(256) void ss_gemm_bf16_wmma(
    const __bf16* __restrict__ A,      // M x K row-major
    const __bf16* __restrict__ W,      // N x K row-major (B^T)
    const float*  __restrict__ bias,   // [N] or nullptr
    void* __restrict__ outv,           // M x N row-major
    int M, int N, int K)
{
    __shared__ alignas(16) __bf16 lds_a[2][64 * LDS_STRIDE];

    const int tid   = threadIdx.x;
    const int lane  = tid & 31;
    const int wave  = tid >> 5;
    const int l15   = lane & 15;
    const int khalf = lane >> 4;

    const int mblock = blockIdx.y * 64;
    const int mwave  = (wave >> 2) * 32;
    const int nbase  = blockIdx.x * 256 + (wave & 3) * 64;

    // A-tile staging assignment: thread -> (row, 8-element chunk)
    const int sr = tid >> 2;           // 0..63
    const int sc = (tid & 3) * 8;      // 0,8,16,24

    v8f acc[2][4];
#pragma unroll
    for (int mi = 0; mi < 2; ++mi)
#pragma unroll
        for (int j = 0; j < 4; ++j)
            acc[mi][j] = v8f{0.f,0.f,0.f,0.f,0.f,0.f,0.f,0.f};

    const int ksteps = K >> 5;

    // prologue: stage K-step 0 into buffer 0
    g2l_16B(&lds_a[0][sr * LDS_STRIDE + sc],
            A + (size_t)(mblock + sr) * K + sc);

    int buf = 0;
    for (int kt = 0; kt < ksteps; ++kt, buf ^= 1) {
        const int kk = kt << 5;
        stage_wait_and_barrier();                      // lds_a[buf] ready

        if (kt + 1 < ksteps)                           // stage next K-step
            g2l_16B(&lds_a[buf ^ 1][sr * LDS_STRIDE + sc],
                    A + (size_t)(mblock + sr) * K + kk + 32 + sc);

        // B operands: issue all 8 b128 loads up front (grouped clause)
        v16bf bv[4];
#pragma unroll
        for (int j = 0; j < 4; ++j) {
            const __bf16* __restrict__ wrow =
                W + (size_t)(nbase + j * 16 + l15) * K + kk + khalf * 8;
            v8bf b0 = *reinterpret_cast<const v8bf*>(wrow);
            v8bf b1 = *reinterpret_cast<const v8bf*>(wrow + 16);
            bv[j] = __builtin_shufflevector(b0, b1,
                0,1,2,3,4,5,6,7,8,9,10,11,12,13,14,15);
        }

        // A operands from LDS (bank-conflict-free: stride 20 dwords, gcd=4, period 16)
        v16bf av[2];
#pragma unroll
        for (int mi = 0; mi < 2; ++mi) {
            const __bf16* arow = &lds_a[buf][(mwave + mi * 16 + l15) * LDS_STRIDE];
            v8bf a0 = *reinterpret_cast<const v8bf*>(arow + khalf * 8);
            v8bf a1 = *reinterpret_cast<const v8bf*>(arow + 16 + khalf * 8);
            av[mi] = __builtin_shufflevector(a0, a1,
                0,1,2,3,4,5,6,7,8,9,10,11,12,13,14,15);
        }

#pragma unroll
        for (int mi = 0; mi < 2; ++mi)
#pragma unroll
            for (int j = 0; j < 4; ++j)
                acc[mi][j] = __builtin_amdgcn_wmma_f32_16x16x32_bf16(
                    false, av[mi], false, bv[j],
                    (short)0, acc[mi][j], false, false);
    }

    // D layout: VGPR v -> row v + 8*khalf, col = l15
#pragma unroll
    for (int mi = 0; mi < 2; ++mi) {
#pragma unroll
        for (int v = 0; v < 8; ++v) {
            const int m = mblock + mwave + mi * 16 + khalf * 8 + v;
#pragma unroll
            for (int j = 0; j < 4; ++j) {
                const int n = nbase + j * 16 + l15;
                float val = acc[mi][j][v];
                size_t idx = (size_t)m * N + n;
                if (EPI == 0) {
                    ((float*)outv)[idx] = val;
                } else if (EPI == 1) {
                    float t = val + bias[n];            // softplus
                    ((float*)outv)[idx] = (t > 20.f) ? t : log1pf(__expf(t));
                } else {
                    ((__bf16*)outv)[idx] = (__bf16)val;
                }
            }
        }
    }
}

// ------------------------------------------------------------------- scan ---
// Wave handles 4 channels. Lane l: channel = base + (l>>3), states (l&7)*4 .. +3.
// ~8 instructions per channel-step (vs 18 for lane-per-state) -> HBM-bound.
__global__ __launch_bounds__(256) void ss_scan_kernel(
    const float*  __restrict__ xin,    // (b, L, d_inner) f32
    const float*  __restrict__ delta,  // (b, L, d_inner) f32 (softplus applied)
    const __bf16* __restrict__ Bm,     // (b, L, d_inner, d_state) bf16
    const __bf16* __restrict__ Cm,     // (b, L, d_inner, d_state) bf16
    const float*  __restrict__ A_log,  // (d_inner, d_state) f32
    __bf16* __restrict__ ybf)          // (b, L, d_inner) bf16, SiLU applied
{
    const int lane = threadIdx.x & 31;
    const int wid  = blockIdx.x * 8 + (threadIdx.x >> 5);   // 0..511
    const int g    = wid * 4;                               // global channel slot
    const int b    = g >> 10;
    const int ch   = (g & (D_INNER - 1)) + (lane >> 3);
    const int sub  = lane & 7;                              // 4-state group (one of 8)
    const float kLog2e = 1.4426950408889634f;

    float a[4], h[4];
#pragma unroll
    for (int i = 0; i < 4; ++i) {
        a[i] = -__expf(A_log[ch * D_STATE + sub * 4 + i]) * kLog2e;
        h[i] = 0.f;
    }

    size_t rbase = (size_t)b * SEQ_LEN * D_INNER + ch;            // += D_INNER / t
    size_t sbase = ((size_t)b * SEQ_LEN * D_INNER + ch) * D_STATE + sub * 4; // += NBC

    for (int t = 0; t < SEQ_LEN; ++t) {
        __builtin_prefetch(Bm + sbase + (size_t)NBC, 0, 0);   // global_prefetch_b8
        __builtin_prefetch(Cm + sbase + (size_t)NBC, 0, 0);

        const float dt = delta[rbase];
        const float xv = xin[rbase];
        v4bf bv = *reinterpret_cast<const v4bf*>(Bm + sbase);
        v4bf cv = *reinterpret_cast<const v4bf*>(Cm + sbase);

        float p = 0.f;
#pragma unroll
        for (int i = 0; i < 4; ++i) {
            h[i] = __builtin_fmaf(__builtin_amdgcn_exp2f(dt * a[i]), h[i],
                                  (float)bv[i] * xv);
            p = __builtin_fmaf((float)cv[i], h[i], p);
        }
        // reduce over the 8 lanes sharing this channel
        p += __shfl_xor(p, 1, 32);
        p += __shfl_xor(p, 2, 32);
        p += __shfl_xor(p, 4, 32);

        if (sub == 0) {
            float s = p / (1.f + __builtin_amdgcn_exp2f(-p * kLog2e));  // SiLU
            ybf[rbase] = (__bf16)s;
        }
        rbase += D_INNER;
        sbase += (size_t)NBC;
    }
}

// ------------------------------------------------------------------- host ---
extern "C" void kernel_launch(void* const* d_in, const int* in_sizes, int n_in,
                              void* d_out, int out_size, void* d_ws, size_t ws_size,
                              hipStream_t stream)
{
    const float* x     = (const float*)d_in[0];  // (2,1024,512)
    const float* Wx    = (const float*)d_in[1];  // (1024,512)
    const float* Wdt   = (const float*)d_in[2];  // (1024,512)
    const float* bdt   = (const float*)d_in[3];  // (1024)
    const float* A_log = (const float*)d_in[4];  // (1024,32)
    const float* WB    = (const float*)d_in[5];  // (32768,512)
    const float* WC    = (const float*)d_in[6];  // (32768,512)
    const float* Wout  = (const float*)d_in[7];  // (512,1024)

    char* p = (char*)d_ws;
    auto carve = [&](size_t bytes) -> void* {
        void* r = (void*)p;
        p += (bytes + 255) & ~(size_t)255;
        return r;
    };
    const size_t n_x     = (size_t)MTOT * D_MODEL;
    const size_t n_wx    = (size_t)D_INNER * D_MODEL;
    const size_t n_wbc   = (size_t)NBC * D_MODEL;
    const size_t n_wout  = (size_t)D_MODEL * D_INNER;
    const size_t n_inner = (size_t)MTOT * D_INNER;
    const size_t n_bc    = (size_t)MTOT * NBC;

    __bf16* xbf    = (__bf16*)carve(n_x * 2);
    __bf16* Wxbf   = (__bf16*)carve(n_wx * 2);
    __bf16* Wdtbf  = (__bf16*)carve(n_wx * 2);
    __bf16* WBbf   = (__bf16*)carve(n_wbc * 2);
    __bf16* WCbf   = (__bf16*)carve(n_wbc * 2);
    __bf16* Woutbf = (__bf16*)carve(n_wout * 2);
    float*  xin    = (float*) carve(n_inner * 4);
    float*  deltap = (float*) carve(n_inner * 4);
    __bf16* Bbf    = (__bf16*)carve(n_bc * 2);
    __bf16* Cbf    = (__bf16*)carve(n_bc * 2);
    __bf16* ybf    = (__bf16*)carve(n_inner * 2);

    auto cvt = [&](const float* in, __bf16* out, size_t n) {
        int blocks = (int)((n / 4 + 255) / 256);
        ss_cvt_f32_to_bf16<<<blocks, 256, 0, stream>>>(in, out, (int)n);
    };
    cvt(x,    xbf,    n_x);
    cvt(Wx,   Wxbf,   n_wx);
    cvt(Wdt,  Wdtbf,  n_wx);
    cvt(WB,   WBbf,   n_wbc);
    cvt(WC,   WCbf,   n_wbc);
    cvt(Wout, Woutbf, n_wout);

    dim3 blk(256);
    ss_gemm_bf16_wmma<0><<<dim3(D_INNER / 256, MTOT / 64), blk, 0, stream>>>(
        xbf, Wxbf, nullptr, (void*)xin, MTOT, D_INNER, D_MODEL);
    ss_gemm_bf16_wmma<1><<<dim3(D_INNER / 256, MTOT / 64), blk, 0, stream>>>(
        xbf, Wdtbf, bdt, (void*)deltap, MTOT, D_INNER, D_MODEL);
    ss_gemm_bf16_wmma<2><<<dim3(NBC / 256, MTOT / 64), blk, 0, stream>>>(
        xbf, WBbf, nullptr, (void*)Bbf, MTOT, NBC, D_MODEL);
    ss_gemm_bf16_wmma<2><<<dim3(NBC / 256, MTOT / 64), blk, 0, stream>>>(
        xbf, WCbf, nullptr, (void*)Cbf, MTOT, NBC, D_MODEL);

    // scan: 512 waves (4 channels each), 8 waves per block
    ss_scan_kernel<<<(BATCH * D_INNER) / 4 / 8, 256, 0, stream>>>(
        xin, deltap, Bbf, Cbf, A_log, ybf);

    ss_gemm_bf16_wmma<0><<<dim3(D_MODEL / 256, MTOT / 64), blk, 0, stream>>>(
        ybf, Woutbf, nullptr, d_out, MTOT, D_MODEL, D_INNER);
}
